// MoEBase_51548197486725
// MI455X (gfx1250) — matrix operand order repaired and checked by reading, not compiled, not run on hardware
//
#include <hip/hip_runtime.h>
#include <hip/hip_bf16.h>

// ---------------- problem constants (match reference) ----------------
#define Tt   2048          // B*S tokens
#define Dd   1024          // in features
#define Ee   16            // routed experts
#define Hh   512           // expert intermediate
#define HS2  2048          // 2*HS (shared fc1 cols)
#define HSs  1024          // shared intermediate
#define H2   1024          // 2*H   (routed fc1 cols)
#define ROUTE_SCALE 1.0f

typedef __attribute__((ext_vector_type(16))) __bf16 v16bf;
typedef __attribute__((ext_vector_type(8)))  float  v8f;

// ---------------- helpers ----------------
__device__ __forceinline__ unsigned short f2bf(float f) {
    union { float f; unsigned u; } v; v.f = f;
    unsigned r = v.u + 0x7FFFu + ((v.u >> 16) & 1u);   // round-to-nearest-even
    return (unsigned short)(r >> 16);
}

__device__ __forceinline__ v8f wmma_bf16(v16bf a, v16bf b, v8f c) {
    return __builtin_amdgcn_wmma_f32_16x16x32_bf16(
        /*neg_a=*/false, a, /*neg_b=*/false, b,
        /*c_mod=*/(short)0, c, /*reuse_a=*/false, /*reuse_b=*/false);
}

// A fragment (16x32 bf16, M x K) gathered from a row-major bf16 matrix.
// rp32 = dword pointer to this lane's row (row = lane&15). ISA layout:
// lanes 0-15 hold K pairs {0,2,4,6,16,18,20,22}; lanes 16-31 the +8 pairs.
__device__ __forceinline__ v16bf load_a32(const unsigned* rp32, int k0) {
    int half = (threadIdx.x >> 4) & 1;
    union { unsigned d[8]; v16bf v; } u;
#pragma unroll
    for (int j = 0; j < 8; ++j) {
        int kk = (j < 4) ? (half * 8 + 2 * j) : (16 + half * 8 + 2 * (j - 4));
        u.d[j] = rp32[(k0 + kk) >> 1];
    }
    return u.v;
}

// B fragment (32x16 bf16, K x N) from pre-packed storage: 1 KB per fragment,
// lane reads 32 contiguous bytes -> two global_load_b128.
__device__ __forceinline__ v16bf load_bp(const uint4* base, size_t fragIdx) {
    int lane = threadIdx.x & 31;
    const uint4* p = base + fragIdx * 64 + (size_t)lane * 2;
    union { uint4 q[2]; v16bf v; } u;
    u.q[0] = p[0]; u.q[1] = p[1];
    return u.v;
}

// ---------------- prep: x -> bf16 pairs ----------------
__global__ void cvt_x_kernel(const float* __restrict__ x, unsigned* __restrict__ xb, long nd) {
    long i0 = (long)blockIdx.x * blockDim.x + threadIdx.x;
    long st = (long)gridDim.x * blockDim.x;
    for (long q = i0; q < nd; q += st) {
        float a = x[2 * q], b = x[2 * q + 1];
        xb[q] = (unsigned)f2bf(a) | ((unsigned)f2bf(b) << 16);
    }
}

// ---------------- prep: pack W[mats][Kd][N] (f32) into B-fragment bf16 ----------------
__global__ void pack_b_kernel(const float* __restrict__ W, unsigned* __restrict__ out,
                              int Kd, int N, long totalDwords) {
    long i0 = (long)blockIdx.x * blockDim.x + threadIdx.x;
    long st = (long)gridDim.x * blockDim.x;
    int numNt = N >> 4;
    long matDwords = (long)Kd * N / 2;
    for (long q = i0; q < totalDwords; q += st) {
        long mat = q / matDwords;
        long r   = q - mat * matDwords;
        int  j   = (int)(r & 7);
        int  l   = (int)((r >> 3) & 31);
        long frag = r >> 8;
        int  nt  = (int)(frag % numNt);
        long kt  = frag / numNt;
        int  n   = (nt << 4) + (l & 15);
        int  k   = (int)(kt << 5) + ((j < 4) ? (((l >> 4) << 3) + (j << 1)) : (16 + ((l >> 4) << 3) + ((j - 4) << 1)));
        const float* src = W + mat * (long)Kd * N;
        float a = src[(long)k * N + n];
        float b = src[(long)(k + 1) * N + n];
        out[q] = (unsigned)f2bf(a) | ((unsigned)f2bf(b) << 16);
    }
}

// ---------------- gate: softmax -> top-2, build per-expert lists ----------------
__global__ void gate_kernel(const float* __restrict__ x, const float* __restrict__ Wg,
                            int* __restrict__ ecnt, int* __restrict__ etok,
                            float* __restrict__ ewt) {
    int wave = threadIdx.x >> 5, lane = threadIdx.x & 31;
    int t = blockIdx.x * 8 + wave;
    const float* xr = x + (size_t)t * Dd;
    float p[Ee];
#pragma unroll
    for (int e = 0; e < Ee; ++e) p[e] = 0.f;
    for (int d = lane; d < Dd; d += 32) {
        float xv = xr[d];
#pragma unroll
        for (int e = 0; e < Ee; ++e) p[e] += xv * Wg[e * Dd + d];
    }
#pragma unroll
    for (int e = 0; e < Ee; ++e) {
        float v = p[e];
        for (int off = 16; off > 0; off >>= 1) v += __shfl_down(v, off, 32);
        p[e] = v;
    }
    if (lane == 0) {
        float mx = p[0];
#pragma unroll
        for (int e = 1; e < Ee; ++e) mx = fmaxf(mx, p[e]);
        float ex[Ee]; float sum = 0.f;
#pragma unroll
        for (int e = 0; e < Ee; ++e) { ex[e] = __expf(p[e] - mx); sum += ex[e]; }
        int i0 = 0;
#pragma unroll
        for (int e = 1; e < Ee; ++e) if (ex[e] > ex[i0]) i0 = e;
        int i1 = (i0 == 0) ? 1 : 0;
#pragma unroll
        for (int e = 0; e < Ee; ++e) if (e != i0 && ex[e] > ex[i1]) i1 = e;
        float inv = 1.f / sum;
        int p0 = atomicAdd(&ecnt[i0], 1);
        etok[i0 * Tt + p0] = t; ewt[i0 * Tt + p0] = ex[i0] * inv * ROUTE_SCALE;
        int p1 = atomicAdd(&ecnt[i1], 1);
        etok[i1 * Tt + p1] = t; ewt[i1 * Tt + p1] = ex[i1] * inv * ROUTE_SCALE;
    }
}

// ---------------- shared expert fc1: hs = u * silu(g), bf16 ----------------
__global__ void shared_fc1_kernel(const unsigned short* __restrict__ xb,
                                  const uint4* __restrict__ Ws1p,
                                  unsigned short* __restrict__ hs) {
    int mt = blockIdx.x;                       // token tile (16 rows)
    int w = threadIdx.x >> 5, lane = threadIdx.x & 31;
    int ntu = blockIdx.y * 8 + w;              // u col tile 0..63  (g tile = +64)
    int m = lane & 15;
    const unsigned* arow = (const unsigned*)(xb + (size_t)(mt * 16 + m) * Dd);
    v8f au = {}, ag = {};
    for (int kt = 0; kt < Dd / 32; ++kt) {
        v16bf a  = load_a32(arow, kt * 32);
        v16bf bu = load_bp(Ws1p, (size_t)kt * (HS2 / 16) + ntu);
        v16bf bg = load_bp(Ws1p, (size_t)kt * (HS2 / 16) + ntu + 64);
        au = wmma_bf16(a, bu, au);
        ag = wmma_bf16(a, bg, ag);
    }
    int mh = (lane >> 4) * 8, n = lane & 15;
#pragma unroll
    for (int i = 0; i < 8; ++i) {
        float u = au[i], g = ag[i];
        float h = u * g / (1.f + __expf(-g));
        hs[(size_t)(mt * 16 + mh + i) * HSs + ntu * 16 + n] = f2bf(h);
    }
}

// ---------------- shared expert fc2: out = hs @ Ws2 (plain store, inits out) ----------------
__global__ void shared_fc2_kernel(const unsigned short* __restrict__ hs,
                                  const uint4* __restrict__ Ws2p,
                                  float* __restrict__ out) {
    int mt = blockIdx.x;
    int w = threadIdx.x >> 5, lane = threadIdx.x & 31;
    int nt = blockIdx.y * 8 + w;               // 0..63 over D
    int m = lane & 15;
    const unsigned* arow = (const unsigned*)(hs + (size_t)(mt * 16 + m) * HSs);
    v8f acc = {};
    for (int kt = 0; kt < HSs / 32; ++kt) {
        v16bf a = load_a32(arow, kt * 32);
        v16bf b = load_bp(Ws2p, (size_t)kt * (Dd / 16) + nt);
        acc = wmma_bf16(a, b, acc);
    }
    int mh = (lane >> 4) * 8, n = lane & 15;
#pragma unroll
    for (int i = 0; i < 8; ++i)
        out[(size_t)(mt * 16 + mh + i) * Dd + nt * 16 + n] = acc[i];
}

// ---------------- routed experts: grouped GEMM + scatter-add ----------------
__global__ void routed_kernel(const unsigned short* __restrict__ xb,
                              const uint4* __restrict__ W1p,
                              const uint4* __restrict__ W2p,
                              const int* __restrict__ ecnt,
                              const int* __restrict__ etok,
                              const float* __restrict__ ewt,
                              float* __restrict__ out) {
    __shared__ int   toks[16];
    __shared__ float wts[16];
    __shared__ unsigned short hbuf[16 * Hh];   // 16 KB bf16 intermediate

    int e = blockIdx.x;
    int tileBase = blockIdx.y * 16;
    int cnt = ecnt[e];
    if (tileBase >= cnt) return;

    if (threadIdx.x < 16) {
        int s = tileBase + threadIdx.x;
        if (s < cnt) { toks[threadIdx.x] = etok[e * Tt + s]; wts[threadIdx.x] = ewt[e * Tt + s]; }
        else         { toks[threadIdx.x] = 0;                wts[threadIdx.x] = 0.f; }
    }
    __syncthreads();

    int w = threadIdx.x >> 5, lane = threadIdx.x & 31;
    int m = lane & 15;
    const unsigned* arow = (const unsigned*)(xb + (size_t)toks[m] * Dd);

    size_t w1base = (size_t)e * (Dd / 32) * (H2 / 16);     // e * 2048 frags
    // fc1 + gated act -> hbuf   (u tiles 0..31, g tile = +32)
    for (int ut = w; ut < Hh / 16; ut += 8) {
        v8f au = {}, ag = {};
        for (int kt = 0; kt < Dd / 32; ++kt) {
            v16bf a  = load_a32(arow, kt * 32);
            v16bf bu = load_bp(W1p, w1base + (size_t)kt * (H2 / 16) + ut);
            v16bf bg = load_bp(W1p, w1base + (size_t)kt * (H2 / 16) + ut + 32);
            au = wmma_bf16(a, bu, au);
            ag = wmma_bf16(a, bg, ag);
        }
        int mh = (lane >> 4) * 8, n = lane & 15;
#pragma unroll
        for (int i = 0; i < 8; ++i) {
            float u = au[i], g = ag[i];
            float h = u * g / (1.f + __expf(-g));
            hbuf[(mh + i) * Hh + ut * 16 + n] = f2bf(h);
        }
    }
    __syncthreads();

    size_t w2base = (size_t)e * (Hh / 32) * (Dd / 16);     // e * 1024 frags
    const unsigned* hrow = (const unsigned*)(&hbuf[m * Hh]);
    // fc2 from LDS, weight by gate, atomically scatter into out
    for (int nt = w; nt < Dd / 16; nt += 8) {
        v8f acc = {};
        for (int kt = 0; kt < Hh / 32; ++kt) {
            v16bf a = load_a32(hrow, kt * 32);
            v16bf b = load_bp(W2p, w2base + (size_t)kt * (Dd / 16) + nt);
            acc = wmma_bf16(a, b, acc);
        }
        int mh = (lane >> 4) * 8, n = lane & 15;
#pragma unroll
        for (int i = 0; i < 8; ++i) {
            int mi = mh + i;
            float v = acc[i] * wts[mi];
            atomicAdd(&out[(size_t)toks[mi] * Dd + nt * 16 + n], v);
        }
    }
}

// ---------------- launcher ----------------
static inline size_t alignup(size_t x) { return (x + 255) & ~(size_t)255; }

extern "C" void kernel_launch(void* const* d_in, const int* in_sizes, int n_in,
                              void* d_out, int out_size, void* d_ws, size_t ws_size,
                              hipStream_t stream) {
    const float* x   = (const float*)d_in[0];   // [T, D]
    const float* Wg  = (const float*)d_in[1];   // [E, D]
    const float* W1  = (const float*)d_in[2];   // [E, D, 2H]
    const float* W2  = (const float*)d_in[3];   // [E, H, D]
    const float* Ws1 = (const float*)d_in[4];   // [D, 2HS]
    const float* Ws2 = (const float*)d_in[5];   // [HS, D]
    float* out = (float*)d_out;

    // workspace layout
    char* ws = (char*)d_ws;
    size_t off = 0;
    int*   ecnt = (int*)(ws + off);            off = alignup(off + Ee * sizeof(int));
    int*   etok = (int*)(ws + off);            off = alignup(off + (size_t)Ee * Tt * sizeof(int));
    float* ewt  = (float*)(ws + off);          off = alignup(off + (size_t)Ee * Tt * sizeof(float));
    unsigned short* xb  = (unsigned short*)(ws + off); off = alignup(off + (size_t)Tt * Dd * 2);
    unsigned short* hs  = (unsigned short*)(ws + off); off = alignup(off + (size_t)Tt * HSs * 2);
    unsigned* W1p  = (unsigned*)(ws + off);    off = alignup(off + (size_t)Ee * Dd * H2 * 2);
    unsigned* W2p  = (unsigned*)(ws + off);    off = alignup(off + (size_t)Ee * Hh * Dd * 2);
    unsigned* Ws1p = (unsigned*)(ws + off);    off = alignup(off + (size_t)Dd * HS2 * 2);
    unsigned* Ws2p = (unsigned*)(ws + off);    off = alignup(off + (size_t)HSs * Dd * 2);
    (void)ws_size; (void)n_in; (void)in_sizes; (void)out_size;

    hipMemsetAsync(ecnt, 0, 256, stream);

    // prep: bf16 conversion + B-fragment packing
    cvt_x_kernel<<<1024, 256, 0, stream>>>(x, (unsigned*)xb, (long)Tt * Dd / 2);
    pack_b_kernel<<<2048, 256, 0, stream>>>(W1,  W1p,  Dd,  H2,  (long)Ee * Dd * H2 / 2);
    pack_b_kernel<<<2048, 256, 0, stream>>>(W2,  W2p,  Hh,  Dd,  (long)Ee * Hh * Dd / 2);
    pack_b_kernel<<<1024, 256, 0, stream>>>(Ws1, Ws1p, Dd,  HS2, (long)Dd * HS2 / 2);
    pack_b_kernel<<<1024, 256, 0, stream>>>(Ws2, Ws2p, HSs, Dd,  (long)HSs * Dd / 2);

    // gate + routing lists
    gate_kernel<<<Tt / 8, 256, 0, stream>>>(x, Wg, ecnt, etok, ewt);

    // shared expert (fc2 plain-stores -> initializes out)
    shared_fc1_kernel<<<dim3(Tt / 16, 8), 256, 0, stream>>>(xb, (const uint4*)Ws1p, hs);
    shared_fc2_kernel<<<dim3(Tt / 16, 8), 256, 0, stream>>>(hs, (const uint4*)Ws2p, out);

    // routed experts (atomic accumulate on top)
    routed_kernel<<<dim3(Ee, Tt / 16), 256, 0, stream>>>(
        xb, (const uint4*)W1p, (const uint4*)W2p, ecnt, etok, ewt, out);
}
